// AttentionBlock_62775241998894
// MI455X (gfx1250) — compile-verified, matrix operand order checked by reading, not run on hardware
//
#include <hip/hip_runtime.h>

// ---- problem constants ----
#define TSEQ 4096
#define NB   8
#define CDIM 512          // C == KEY_SIZE == V == 512
#define MTOT (NB * TSEQ)  // 32768 rows for projections

typedef __bf16 bf16_t;
typedef __attribute__((ext_vector_type(16))) __bf16 v16bf;
typedef __attribute__((ext_vector_type(8)))  float  v8f;

#define WMMA_BF16(a, b, c) \
  __builtin_amdgcn_wmma_f32_16x16x32_bf16(false, (a), false, (b), (short)0, (c), false, false)

// ---- CDNA5 async global->LDS copy (ASYNCcnt-tracked, no VGPR round trip) ----
// VDST = VGPR with LDS byte offset, VADDR = 64-bit global address (GV mode).
__device__ __forceinline__ void ld_async_b128(bf16_t* lds_dst, const bf16_t* gsrc) {
  unsigned loff = (unsigned)(uintptr_t)lds_dst;  // low 32 bits of flat ptr = LDS offset
  asm volatile("global_load_async_to_lds_b128 %0, %1, off"
               :
               : "v"(loff), "v"(gsrc)
               : "memory");
}
__device__ __forceinline__ void wait_async0() {
  asm volatile("s_wait_asynccnt 0x0" ::: "memory");
}

// Load one 16x32(bf16) A/B fragment for v_wmma_f32_16x16x32_bf16.
// p points at (row_ptr + half*8); dword v holds K pair ((v&4)<<2)+((v&3)<<1).
__device__ __forceinline__ v16bf frag_ld(const bf16_t* p) {
  v16bf f;
#pragma unroll
  for (int v = 0; v < 8; ++v) {
    const int k = ((v & 4) << 2) + ((v & 3) << 1);
    f[2 * v]     = p[k];
    f[2 * v + 1] = p[k + 1];
  }
  return f;
}

// issue async copies for a 128x32 bf16 tile (row stride 512 in global, 40 in LDS)
__device__ __forceinline__ void stage128x32_async(bf16_t* buf, const bf16_t* src, int t) {
#pragma unroll
  for (int it = 0; it < 2; ++it) {
    int tt = t + 256 * it, row = tt >> 2, seg = tt & 3;
    ld_async_b128(buf + row * 40 + seg * 8, src + (size_t)row * 512 + seg * 8);
  }
}

// ---------------- kernel 0: f32 -> bf16 convert (weights) ----------------
__global__ void cvt_bf16_kernel(const float* __restrict__ src, bf16_t* __restrict__ dst, int n) {
  int i = blockIdx.x * blockDim.x + threadIdx.x;
  int stride = gridDim.x * blockDim.x;
  for (; i < n; i += stride) dst[i] = (bf16_t)src[i];
}

// ---------------- kernel 1: x[B,C,T] f32 -> xt[B*T,C] bf16 (transpose) ----------------
__global__ void transpose_cvt_kernel(const float* __restrict__ x, bf16_t* __restrict__ xt) {
  __shared__ float tile[32][33];
  const int t0 = blockIdx.x * 32, c0 = blockIdx.y * 32, b = blockIdx.z;
  const int tx = threadIdx.x, ty = threadIdx.y;  // 32 x 8
  const float* xp = x + (size_t)b * CDIM * TSEQ;
#pragma unroll
  for (int i = 0; i < 4; ++i) {
    int c = c0 + ty + 8 * i;
    tile[ty + 8 * i][tx] = xp[(size_t)c * TSEQ + t0 + tx];  // coalesced along t
  }
  __syncthreads();
  bf16_t* yp = xt + (size_t)b * TSEQ * CDIM;
#pragma unroll
  for (int i = 0; i < 4; ++i) {
    int t = t0 + ty + 8 * i;
    yp[(size_t)t * CDIM + c0 + tx] = (bf16_t)tile[tx][ty + 8 * i];  // coalesced along c
  }
}

// ---------------- kernel 2: copy x into out[:, 0:512, :] ----------------
__global__ void copy_x_kernel(const float4* __restrict__ x4, float4* __restrict__ out4) {
  const int per4 = CDIM * TSEQ / 4;  // float4 per batch of x
  const int n4 = NB * per4;
  int i = blockIdx.x * blockDim.x + threadIdx.x;
  int stride = gridDim.x * blockDim.x;
  for (; i < n4; i += stride) {
    int b = i / per4, r = i - b * per4;
    out4[(size_t)b * (2 * per4) + r] = x4[i];  // out channel dim = 1024
  }
}

// ---------------- kernel 3: Y[M,512] = xt[M,512] @ W[512,512]^T + bias ----------------
// 128(M) x 64(N) tile, 8 waves as 4x2 (wave = 32x32 = 2x2 WMMA tiles),
// double-buffered async-LDS staging pipelined against WMMA.
// Double buffers are selected with an integer element-offset toggle (NOT a
// pointer select) so address-space inference keeps all tile reads on ds_load.
__global__ void __launch_bounds__(256) proj_kernel(const bf16_t* __restrict__ xt,
                                                   const bf16_t* __restrict__ W,
                                                   const float* __restrict__ bias,
                                                   bf16_t* __restrict__ Y) {
  extern __shared__ char smem[];
  bf16_t* lds_a = (bf16_t*)smem;           // 2 x 128x40 (elem offsets 0 / 5120)
  bf16_t* lds_b = (bf16_t*)smem + 10240;   // 2 x  64x40 (elem offsets 0 / 2560)
  const int m0 = blockIdx.x * 128, n0 = blockIdx.y * 64;
  const int t = threadIdx.x, wave = t >> 5, lane = t & 31;
  const int wm = wave >> 1, wn = wave & 1;
  const int half = lane >> 4, r = lane & 15;

  // prologue: chunk 0 into buffer 0
  stage128x32_async(lds_a, xt + (size_t)m0 * CDIM, t);
  {
    int row = t >> 2, seg = t & 3;
    ld_async_b128(lds_b + row * 40 + seg * 8, W + (size_t)(n0 + row) * CDIM + seg * 8);
  }

  v8f acc[2][2] = {};
  int ao = 0, bo = 0;  // element-offset toggles
  for (int kc = 0; kc < CDIM; kc += 32, ao ^= 5120, bo ^= 2560) {
    wait_async0();
    __syncthreads();
    if (kc + 32 < CDIM) {  // prefetch next chunk into the other buffer
      stage128x32_async(lds_a + (ao ^ 5120), xt + (size_t)m0 * CDIM + kc + 32, t);
      int row = t >> 2, seg = t & 3;
      ld_async_b128(lds_b + (bo ^ 2560) + row * 40 + seg * 8,
                    W + (size_t)(n0 + row) * CDIM + kc + 32 + seg * 8);
    }
    v16bf a0 = frag_ld(lds_a + ao + (wm * 32 + r) * 40 + half * 8);
    v16bf a1 = frag_ld(lds_a + ao + (wm * 32 + 16 + r) * 40 + half * 8);
    v16bf b0 = frag_ld(lds_b + bo + (wn * 32 + r) * 40 + half * 8);
    v16bf b1 = frag_ld(lds_b + bo + (wn * 32 + 16 + r) * 40 + half * 8);
    acc[0][0] = WMMA_BF16(a0, b0, acc[0][0]);
    acc[0][1] = WMMA_BF16(a0, b1, acc[0][1]);
    acc[1][0] = WMMA_BF16(a1, b0, acc[1][0]);
    acc[1][1] = WMMA_BF16(a1, b1, acc[1][1]);
  }
#pragma unroll
  for (int ti = 0; ti < 2; ++ti)
#pragma unroll
    for (int tj = 0; tj < 2; ++tj) {
      int n = n0 + wn * 32 + tj * 16 + r;
      float bv = bias[n];
#pragma unroll
      for (int d = 0; d < 8; ++d) {
        int m = m0 + wm * 32 + ti * 16 + half * 8 + d;
        Y[(size_t)m * 512 + n] = (bf16_t)(acc[ti][tj][d] + bv);
      }
    }
}

// ---------------- kernel 4: colsum[j] = sum_{i>=j} exp(q_i . k_j / sqrt(K)) ----------------
__global__ void __launch_bounds__(256) colsum_kernel(const bf16_t* __restrict__ qb,
                                                     const bf16_t* __restrict__ kb,
                                                     float* __restrict__ colsum) {
  extern __shared__ char smem[];
  bf16_t* lds_q = (bf16_t*)smem;           // 2 x 128x40 (offsets 0 / 5120)
  bf16_t* lds_k = (bf16_t*)smem + 10240;   // 2 x 128x40 (offsets 0 / 5120)
  float* colacc = (float*)(smem + 40960);  // 128 floats
  const int j0 = blockIdx.x * 128, b = blockIdx.y;
  const int t = threadIdx.x, wave = t >> 5, lane = t & 31;
  const int wm = wave >> 1, wn = wave & 1;  // 4(i) x 2(j) waves; wave = 32(i) x 64(j)
  const int half = lane >> 4, r = lane & 15;
  const bf16_t* qp = qb + (size_t)b * TSEQ * 512;
  const bf16_t* kp = kb + (size_t)b * TSEQ * 512;
  const float scale = 0.044194173824159216f;  // 1/sqrt(512)
  if (t < 128) colacc[t] = 0.f;

  for (int i0 = j0; i0 < TSEQ; i0 += 128) {
    v8f s[2][4] = {};
    stage128x32_async(lds_q, qp + (size_t)i0 * 512, t);
    stage128x32_async(lds_k, kp + (size_t)j0 * 512, t);
    int po = 0;
    for (int kc = 0; kc < 512; kc += 32, po ^= 5120) {
      wait_async0();
      __syncthreads();
      if (kc + 32 < 512) {
        stage128x32_async(lds_q + (po ^ 5120), qp + (size_t)i0 * 512 + kc + 32, t);
        stage128x32_async(lds_k + (po ^ 5120), kp + (size_t)j0 * 512 + kc + 32, t);
      }
      v16bf a0 = frag_ld(lds_q + po + (wm * 32 + r) * 40 + half * 8);
      v16bf a1 = frag_ld(lds_q + po + (wm * 32 + 16 + r) * 40 + half * 8);
#pragma unroll
      for (int tj = 0; tj < 4; ++tj) {
        v16bf bb = frag_ld(lds_k + po + (wn * 64 + tj * 16 + r) * 40 + half * 8);
        s[0][tj] = WMMA_BF16(a0, bb, s[0][tj]);
        s[1][tj] = WMMA_BF16(a1, bb, s[1][tj]);
      }
    }
#pragma unroll
    for (int ti = 0; ti < 2; ++ti)
#pragma unroll
      for (int tj = 0; tj < 4; ++tj) {
        int jl = wn * 64 + tj * 16 + r;
        int j = j0 + jl;
        float part = 0.f;
#pragma unroll
        for (int d = 0; d < 8; ++d) {
          int i = i0 + wm * 32 + ti * 16 + half * 8 + d;
          part += (j <= i) ? __expf(s[ti][tj][d] * scale) : 0.f;
        }
        atomicAdd(&colacc[jl], part);
      }
  }
  __syncthreads();
  if (t < 128) colsum[(size_t)b * TSEQ + j0 + t] = colacc[t];
}

// ---------------- kernel 5: vhat[j,:] = v[j,:] / colsum[j] (in place, bf16) ----------------
__global__ void scale_v_kernel(bf16_t* __restrict__ vb, const float* __restrict__ colsum) {
  const int n = MTOT * 512;
  int i = blockIdx.x * blockDim.x + threadIdx.x;
  int stride = gridDim.x * blockDim.x;
  for (; i < n; i += stride) {
    int row = i >> 9;
    vb[i] = (bf16_t)((float)vb[i] / colsum[row]);
  }
}

// ---------------- kernel 6: act = E @ vhat, written transposed to out[:,512:,:] ----------------
__global__ void __launch_bounds__(256) attnv_kernel(const bf16_t* __restrict__ qb,
                                                    const bf16_t* __restrict__ kb,
                                                    const bf16_t* __restrict__ vb,
                                                    float* __restrict__ out) {
  extern __shared__ char smem[];
  bf16_t* lds_q = (bf16_t*)smem;           // 2 x 128x40 (offsets 0 / 5120)
  bf16_t* lds_k = (bf16_t*)smem + 10240;   // 2 x 128x40 (offsets 0 / 5120)
  bf16_t* lds_e = (bf16_t*)smem + 20480;   // 128x136 bf16 = 34816 B
  bf16_t* lds_v = (bf16_t*)smem + 37888;   // 256x136 bf16 = 69632 B (transposed: [v][j])
  const int i0 = blockIdx.x * 128, v0 = blockIdx.y * 256, b = blockIdx.z;
  const int t = threadIdx.x, wave = t >> 5, lane = t & 31;
  const int half = lane >> 4, r = lane & 15;
  const int swm = wave >> 1, swn = wave & 1;  // score waves: 4(i) x 2(j)
  const int awm = wave >> 1, awv = wave & 1;  // AV waves:    4(i) x 2(v), wave = 32 x 128
  const bf16_t* qp = qb + (size_t)b * TSEQ * 512;
  const bf16_t* kp = kb + (size_t)b * TSEQ * 512;
  const bf16_t* vp = vb + (size_t)b * TSEQ * 512;
  const float scale = 0.044194173824159216f;

  v8f acc[2][8] = {};
  for (int j0 = 0; j0 <= i0; j0 += 128) {
    // --- phase 1: E tile (128x128), double-buffered async staging ---
    v8f s[2][4] = {};
    stage128x32_async(lds_q, qp + (size_t)i0 * 512, t);
    stage128x32_async(lds_k, kp + (size_t)j0 * 512, t);
    int po = 0;
    for (int kc = 0; kc < 512; kc += 32, po ^= 5120) {
      wait_async0();
      __syncthreads();
      if (kc + 32 < 512) {
        stage128x32_async(lds_q + (po ^ 5120), qp + (size_t)i0 * 512 + kc + 32, t);
        stage128x32_async(lds_k + (po ^ 5120), kp + (size_t)j0 * 512 + kc + 32, t);
      }
      v16bf a0 = frag_ld(lds_q + po + (swm * 32 + r) * 40 + half * 8);
      v16bf a1 = frag_ld(lds_q + po + (swm * 32 + 16 + r) * 40 + half * 8);
#pragma unroll
      for (int tj = 0; tj < 4; ++tj) {
        v16bf bb = frag_ld(lds_k + po + (swn * 64 + tj * 16 + r) * 40 + half * 8);
        s[0][tj] = WMMA_BF16(a0, bb, s[0][tj]);
        s[1][tj] = WMMA_BF16(a1, bb, s[1][tj]);
      }
    }
#pragma unroll
    for (int ti = 0; ti < 2; ++ti)
#pragma unroll
      for (int tj = 0; tj < 4; ++tj) {
        int jl = swn * 64 + tj * 16 + r, j = j0 + jl;
#pragma unroll
        for (int d = 0; d < 8; ++d) {
          int il = swm * 32 + ti * 16 + half * 8 + d;
          float e = (j <= i0 + il) ? __expf(s[ti][tj][d] * scale) : 0.f;
          lds_e[il * 136 + jl] = (bf16_t)e;
        }
      }
    __syncthreads();
    // --- phase 2: stage vhat[j0:j0+128, v0:v0+256] transposed: lds_v[v][j] ---
#pragma unroll
    for (int seg = 0; seg < 16; ++seg) {
      union { uint4 u; bf16_t h[8]; } tmp;
#pragma unroll
      for (int jj = 0; jj < 8; ++jj)  // coalesced along v across the 256 threads
        tmp.h[jj] = vp[(size_t)(j0 + seg * 8 + jj) * 512 + v0 + t];
      *(uint4*)(lds_v + t * 136 + seg * 8) = tmp.u;
    }
    __syncthreads();
    // --- phase 3: acc += E @ vhat ---
#pragma unroll
    for (int jk = 0; jk < 4; ++jk) {
      v16bf a0 = frag_ld(lds_e + (awm * 32 + r) * 136 + jk * 32 + half * 8);
      v16bf a1 = frag_ld(lds_e + (awm * 32 + 16 + r) * 136 + jk * 32 + half * 8);
#pragma unroll
      for (int tv = 0; tv < 8; ++tv) {
        v16bf bb = frag_ld(lds_v + (awv * 128 + tv * 16 + r) * 136 + jk * 32 + half * 8);
        acc[0][tv] = WMMA_BF16(a0, bb, acc[0][tv]);
        acc[1][tv] = WMMA_BF16(a1, bb, acc[1][tv]);
      }
    }
  }
  // write act transposed: out[b][512 + v][i]
  float* op = out + (size_t)b * 1024 * TSEQ;
#pragma unroll
  for (int ti = 0; ti < 2; ++ti)
#pragma unroll
    for (int tv = 0; tv < 8; ++tv) {
      int vg = 512 + v0 + awv * 128 + tv * 16 + r;
#pragma unroll
      for (int d = 0; d < 8; ++d) {
        int ig = i0 + awm * 32 + ti * 16 + half * 8 + d;
        op[(size_t)vg * TSEQ + ig] = acc[ti][tv][d];
      }
    }
}

extern "C" void kernel_launch(void* const* d_in, const int* in_sizes, int n_in,
                              void* d_out, int out_size, void* d_ws, size_t ws_size,
                              hipStream_t stream) {
  const float* x  = (const float*)d_in[0];
  const float* Wq = (const float*)d_in[1];
  const float* bq = (const float*)d_in[2];
  const float* Wk = (const float*)d_in[3];
  const float* bk = (const float*)d_in[4];
  const float* Wv = (const float*)d_in[5];
  const float* bv = (const float*)d_in[6];
  float* out = (float*)d_out;

  char* ws = (char*)d_ws;
  size_t off = 0;
  auto alloc = [&](size_t bytes) -> void* {
    void* p = ws + off;
    off = (off + bytes + 255) & ~(size_t)255;
    return p;
  };
  bf16_t* xt = (bf16_t*)alloc((size_t)MTOT * 512 * 2);  // 32 MB
  bf16_t* wq = (bf16_t*)alloc(512 * 512 * 2);
  bf16_t* wk = (bf16_t*)alloc(512 * 512 * 2);
  bf16_t* wv = (bf16_t*)alloc(512 * 512 * 2);
  bf16_t* qbuf = (bf16_t*)alloc((size_t)MTOT * 512 * 2);  // 32 MB
  bf16_t* kbuf = (bf16_t*)alloc((size_t)MTOT * 512 * 2);  // 32 MB
  bf16_t* vbuf = (bf16_t*)alloc((size_t)MTOT * 512 * 2);  // 32 MB
  float* colsum = (float*)alloc((size_t)MTOT * 4);

  // 1) weights -> bf16
  cvt_bf16_kernel<<<512, 256, 0, stream>>>(Wq, wq, 512 * 512);
  cvt_bf16_kernel<<<512, 256, 0, stream>>>(Wk, wk, 512 * 512);
  cvt_bf16_kernel<<<512, 256, 0, stream>>>(Wv, wv, 512 * 512);
  // 2) x -> xt (transposed bf16); x -> out[:, :512, :]
  transpose_cvt_kernel<<<dim3(TSEQ / 32, CDIM / 32, NB), dim3(32, 8), 0, stream>>>(x, xt);
  copy_x_kernel<<<4096, 256, 0, stream>>>((const float4*)x, (float4*)out);
  // 3) QKV projections (WMMA, async double-buffered staging)
  proj_kernel<<<dim3(MTOT / 128, 512 / 64), 256, 30720, stream>>>(xt, wq, bq, qbuf);
  proj_kernel<<<dim3(MTOT / 128, 512 / 64), 256, 30720, stream>>>(xt, wk, bk, kbuf);
  proj_kernel<<<dim3(MTOT / 128, 512 / 64), 256, 30720, stream>>>(xt, wv, bv, vbuf);
  // 4) per-key-column exp sums (WMMA score recompute)
  colsum_kernel<<<dim3(TSEQ / 128, NB), 256, 41472, stream>>>(qbuf, kbuf, colsum);
  // 5) fold normalizer into V
  scale_v_kernel<<<8192, 256, 0, stream>>>(vbuf, colsum);
  // 6) act = E @ vhat (WMMA score recompute + AV), write transposed
  attnv_kernel<<<dim3(TSEQ / 128, 512 / 256, NB), 256, 145408, stream>>>(qbuf, kbuf, vbuf, out);
}
